// PSABlock_47390669144858
// MI455X (gfx1250) — matrix operand order repaired and checked by reading, not compiled, not run on hardware
//
#include <hip/hip_runtime.h>
#include <hip/hip_bf16.h>
#include <stdint.h>

typedef __attribute__((ext_vector_type(16))) __bf16       v16bf;
typedef __attribute__((ext_vector_type(8)))  float        v8f;
typedef __attribute__((ext_vector_type(4)))  unsigned int u32x4;
typedef __attribute__((ext_vector_type(2)))  unsigned int u32x2;
typedef __attribute__((ext_vector_type(2)))  float        f32x2;
typedef __attribute__((ext_vector_type(4)))  float        f32x4;
typedef __attribute__((ext_vector_type(2)))  __bf16       bf16x2;
typedef __attribute__((ext_vector_type(4)))  __bf16       bf16x4;

#define B_   8
#define C_   256
#define H_   80
#define W_   80
#define HW   6400
#define EPSF 1e-5f

// one v_cvt_pk_bf16_f32: packs (a,b) -> bf16 pair, a in low half
__device__ __forceinline__ unsigned int pk_bf16x2(float a, float b) {
    f32x2 f; f.x = a; f.y = b;
    union { bf16x2 v; unsigned int u; } p;
    p.v = __builtin_convertvector(f, bf16x2);
    return p.u;
}

// ---------------------------------------------------------------------------
// WMMA GEMM: out[b,o,m] = epi( sum_k W[o,k] * X[b,k,m] )
//   A (WMMA M-dim) = weight rows (o), B (WMMA N-dim) = spatial (m) -> stores
//   along m are contiguous across lanes. Block tile 128(o) x 128(m), 8 waves,
//   each wave 64(o) x 32(m) = 4x2 v_wmma_f32_16x16x32_bf16 per K-step of 32.
//   Register double-buffering: next K-tile's global loads are issued before
//   the WMMA compute of the current tile, hiding HBM latency.
// epilogue: y = acc*scale[o]+shift[o] (or acc+shift if scale==null);
//           optional SiLU; optional += res[b,o,m]
// grid: (HW/128, O/128, B), block 256
// ---------------------------------------------------------------------------
__global__ __launch_bounds__(256)
void gemm_wx(const float* __restrict__ X, const float* __restrict__ W,
             const float* __restrict__ scale, const float* __restrict__ shift,
             const float* __restrict__ res, float* __restrict__ out,
             int K, int O, int do_silu)
{
    __shared__ __bf16 Asb[128 * 32]; // W tile [o][k], k contiguous
    __shared__ __bf16 Bsb[128 * 32]; // X tile [m][k], k contiguous

    const int t    = threadIdx.x;
    const int lane = t & 31;
    const int wid  = t >> 5;
    const int wo   = wid & 1;    // o half   (0/1)  -> +64
    const int wm   = wid >> 1;   // m quarter(0..3) -> +32
    const int lh   = lane >> 4;  // lane half
    const int l16  = lane & 15;

    const int mBase = blockIdx.x * 128;
    const int oBase = blockIdx.y * 128;
    const int b     = blockIdx.z;

    const float* Xb = X + (size_t)b * K * HW;

    // ---- per-thread staging coordinates (fixed across K-steps) ------------
    // W tile: 4 units, unit = float4 along k of one o row
    int oA[4], k4A[4];
#pragma unroll
    for (int i = 0; i < 4; ++i) {
        int it = i * 256 + t;
        oA[i]  = it >> 3;
        k4A[i] = (it & 7) << 2;
    }
    // X tile: 2 units, unit = float4 along m from two adjacent k rows
    // (pack (k,k+1) pairs -> one ds_store_b32 per m, transposed layout)
    int kpB[2], m4B[2];
#pragma unroll
    for (int i = 0; i < 2; ++i) {
        int u  = i * 256 + t;
        kpB[i] = (u >> 5) << 1;      // even k row index
        m4B[i] = (u & 31) << 2;
    }

    v8f acc[4][2];
#pragma unroll
    for (int i = 0; i < 4; ++i)
#pragma unroll
        for (int j = 0; j < 2; ++j) acc[i][j] = v8f{};

    float4 wv[4], xlo[2], xhi[2];
    // prime the register buffer with K-tile 0
#pragma unroll
    for (int i = 0; i < 4; ++i)
        wv[i] = *(const float4*)(W + (size_t)(oBase + oA[i]) * K + k4A[i]);
#pragma unroll
    for (int i = 0; i < 2; ++i) {
        xlo[i] = *(const float4*)(Xb + (size_t)(kpB[i] + 0) * HW + mBase + m4B[i]);
        xhi[i] = *(const float4*)(Xb + (size_t)(kpB[i] + 1) * HW + mBase + m4B[i]);
    }

    for (int kt = 0; kt < K; kt += 32) {
        // ---- drain register buffer into LDS (packed bf16 converts) --------
#pragma unroll
        for (int i = 0; i < 4; ++i) {
            f32x4 f; f.x = wv[i].x; f.y = wv[i].y; f.z = wv[i].z; f.w = wv[i].w;
            union { bf16x4 v; u32x2 u; } p;
            p.v = __builtin_convertvector(f, bf16x4);
            *(u32x2*)(&Asb[oA[i] * 32 + k4A[i]]) = p.u;
        }
#pragma unroll
        for (int i = 0; i < 2; ++i) {
            *(unsigned int*)(&Bsb[(m4B[i] + 0) * 32 + kpB[i]]) = pk_bf16x2(xlo[i].x, xhi[i].x);
            *(unsigned int*)(&Bsb[(m4B[i] + 1) * 32 + kpB[i]]) = pk_bf16x2(xlo[i].y, xhi[i].y);
            *(unsigned int*)(&Bsb[(m4B[i] + 2) * 32 + kpB[i]]) = pk_bf16x2(xlo[i].z, xhi[i].z);
            *(unsigned int*)(&Bsb[(m4B[i] + 3) * 32 + kpB[i]]) = pk_bf16x2(xlo[i].w, xhi[i].w);
        }
        __syncthreads();

        // ---- issue next K-tile's global loads (overlap with WMMA below) ---
        if (kt + 32 < K) {
#pragma unroll
            for (int i = 0; i < 4; ++i)
                wv[i] = *(const float4*)(W + (size_t)(oBase + oA[i]) * K + kt + 32 + k4A[i]);
#pragma unroll
            for (int i = 0; i < 2; ++i) {
                xlo[i] = *(const float4*)(Xb + (size_t)(kt + 32 + kpB[i] + 0) * HW + mBase + m4B[i]);
                xhi[i] = *(const float4*)(Xb + (size_t)(kt + 32 + kpB[i] + 1) * HW + mBase + m4B[i]);
            }
            if (kt + 64 < K) { // warm L2 one more tile ahead (global_prefetch_b8)
                __builtin_prefetch(W  + (size_t)(oBase + oA[0]) * K + kt + 64 + k4A[0], 0, 1);
                __builtin_prefetch(Xb + (size_t)(kt + 64 + kpB[0]) * HW + mBase + m4B[0], 0, 1);
            }
        }

        // ---- fragments per ISA layouts ------------------------------------
        // A 16x32 bf16: lanes0-15 K{0-7,16-23}, lanes16-31 K{8-15,24-31}
        // B 32x16 bf16: lanes0-15 K{0-7,16-23}, lanes16-31 K{8-15,24-31}
        union { v16bf v; u32x4 u[2]; } afr[4], bfr[2];
#pragma unroll
        for (int oi = 0; oi < 4; ++oi) {
            int orow = wo * 64 + oi * 16 + l16;
            int ka   = lh * 8;
            afr[oi].u[0] = *(const u32x4*)(&Asb[orow * 32 + ka]);
            afr[oi].u[1] = *(const u32x4*)(&Asb[orow * 32 + ka + 16]);
        }
#pragma unroll
        for (int mi = 0; mi < 2; ++mi) {
            int mcol = wm * 32 + mi * 16 + l16;
            int kb   = lh * 8;
            bfr[mi].u[0] = *(const u32x4*)(&Bsb[mcol * 32 + kb]);
            bfr[mi].u[1] = *(const u32x4*)(&Bsb[mcol * 32 + kb + 16]);
        }
#pragma unroll
        for (int oi = 0; oi < 4; ++oi)
#pragma unroll
            for (int mi = 0; mi < 2; ++mi)
                acc[oi][mi] = __builtin_amdgcn_wmma_f32_16x16x32_bf16(
                    false, afr[oi].v, false, bfr[mi].v,
                    (short)0, acc[oi][mi], false, false);

        __syncthreads(); // LDS consumed; safe to overwrite next iteration
    }

    // ---- epilogue: D layout VGPR r -> o_local = r + 8*laneHalf, n = lane&15
#pragma unroll
    for (int oi = 0; oi < 4; ++oi) {
#pragma unroll
        for (int mi = 0; mi < 2; ++mi) {
#pragma unroll
            for (int r = 0; r < 8; ++r) {
                int o = oBase + wo * 64 + oi * 16 + lh * 8 + r;
                int m = mBase + wm * 32 + mi * 16 + l16;
                float y = acc[oi][mi][r];
                if (scale) y = y * scale[o] + shift[o];
                else       y = y + shift[o];
                if (do_silu) y = y * (1.0f / (1.0f + __expf(-y)));
                size_t idx = ((size_t)b * O + o) * HW + m;
                if (res) y += res[idx];
                out[idx] = y;
            }
        }
    }
}

// ---------------------------------------------------------------------------
// Dilated 3x3 neighborhood attention, one head of 64 ch per dilation branch.
// Zero-padded neighbors participate in softmax with logit 0 (matches unfold
// zero padding in the reference). Thread-per-pixel, coalesced along spatial.
// grid: (HW/256, 4, B)
// ---------------------------------------------------------------------------
__global__ __launch_bounds__(256)
void attn_kernel(const float* __restrict__ qkv, float* __restrict__ stacked)
{
    const int n  = blockIdx.x * 256 + threadIdx.x;
    const int di = blockIdx.y;
    const int b  = blockIdx.z;
    const int r  = 1 << di;           // DIL = 1,2,4,8
    const int y  = n / W_;
    const int x  = n - y * W_;

    const float* q = qkv + ((size_t)b * 768 + 0   + di * 64) * HW;
    const float* k = qkv + ((size_t)b * 768 + 256 + di * 64) * HW;
    const float* v = qkv + ((size_t)b * 768 + 512 + di * 64) * HW;

    int  off[9];
    bool val[9];
#pragma unroll
    for (int j = 0; j < 9; ++j) {
        int dy = (j / 3 - 1) * r;
        int dx = (j % 3 - 1) * r;
        int yy = y + dy, xx = x + dx;
        val[j] = (yy >= 0) && (yy < H_) && (xx >= 0) && (xx < W_);
        off[j] = dy * W_ + dx;
    }

    float acc[9];
#pragma unroll
    for (int j = 0; j < 9; ++j) acc[j] = 0.f;

    for (int c = 0; c < 64; ++c) {
        const float* kc = k + (size_t)c * HW + n;
        float qc = q[(size_t)c * HW + n];
#pragma unroll
        for (int j = 0; j < 9; ++j)
            acc[j] += qc * (val[j] ? kc[off[j]] : 0.f);
    }

    float mx = -1e30f;
#pragma unroll
    for (int j = 0; j < 9; ++j) { acc[j] *= 0.125f; mx = fmaxf(mx, acc[j]); }
    float p[9], s = 0.f;
#pragma unroll
    for (int j = 0; j < 9; ++j) { p[j] = __expf(acc[j] - mx); s += p[j]; }
    float inv = 1.f / s;
#pragma unroll
    for (int j = 0; j < 9; ++j) p[j] *= inv;

    float* outp = stacked + (((size_t)b * 4 + di) * 64) * HW + n;
    for (int c = 0; c < 64; ++c) {
        const float* vc = v + (size_t)c * HW + n;
        float o = 0.f;
#pragma unroll
        for (int j = 0; j < 9; ++j)
            o += p[j] * (val[j] ? vc[off[j]] : 0.f);
        outp[(size_t)c * HW] = o;
    }
}

// ---------------------------------------------------------------------------
// fused = fc_w @ stacked + fc_b + stacked ; LayerNorm over 64 ch per branch;
// writes normed as [B, C, HW] so the proj GEMM consumes it directly.
// Two-pass (mean/var then normalize) to keep registers small.
// grid: (HW/256, 1, B)
// ---------------------------------------------------------------------------
__global__ __launch_bounds__(256)
void fuse_ln_kernel(const float* __restrict__ stacked,
                    const float* __restrict__ fc_w, const float* __restrict__ fc_b,
                    const float* __restrict__ ln_g, const float* __restrict__ ln_b,
                    float* __restrict__ normed)
{
    const int n = blockIdx.x * 256 + threadIdx.x;
    const int b = blockIdx.z;
    const float* st = stacked + (size_t)b * 4 * 64 * HW;

    float fw[16], fb[4];
#pragma unroll
    for (int i = 0; i < 16; ++i) fw[i] = fc_w[i];
#pragma unroll
    for (int i = 0; i < 4; ++i) fb[i] = fc_b[i];

#pragma unroll
    for (int tt = 0; tt < 4; ++tt) {
        float sum = 0.f, sq = 0.f;
        for (int c = 0; c < 64; ++c) {
            float s0 = st[(size_t)(0 * 64 + c) * HW + n];
            float s1 = st[(size_t)(1 * 64 + c) * HW + n];
            float s2 = st[(size_t)(2 * 64 + c) * HW + n];
            float s3 = st[(size_t)(3 * 64 + c) * HW + n];
            float f  = fb[tt] + fw[tt*4+0]*s0 + fw[tt*4+1]*s1
                              + fw[tt*4+2]*s2 + fw[tt*4+3]*s3;
            f += (tt == 0) ? s0 : (tt == 1) ? s1 : (tt == 2) ? s2 : s3;
            sum += f; sq += f * f;
        }
        float mu   = sum * (1.f / 64.f);
        float var  = sq  * (1.f / 64.f) - mu * mu;
        float rinv = rsqrtf(var + EPSF);
        for (int c = 0; c < 64; ++c) {
            float s0 = st[(size_t)(0 * 64 + c) * HW + n];
            float s1 = st[(size_t)(1 * 64 + c) * HW + n];
            float s2 = st[(size_t)(2 * 64 + c) * HW + n];
            float s3 = st[(size_t)(3 * 64 + c) * HW + n];
            float f  = fb[tt] + fw[tt*4+0]*s0 + fw[tt*4+1]*s1
                              + fw[tt*4+2]*s2 + fw[tt*4+3]*s3;
            f += (tt == 0) ? s0 : (tt == 1) ? s1 : (tt == 2) ? s2 : s3;
            float yv = (f - mu) * rinv * ln_g[c] + ln_b[c];
            normed[((size_t)b * C_ + tt * 64 + c) * HW + n] = yv;
        }
    }
}

// ---------------------------------------------------------------------------
// Fold BN params into per-channel scale/shift: a = g*rsqrt(v+eps), s = b-m*a
// ---------------------------------------------------------------------------
__global__ void bn_prep(const float* g1, const float* b1, const float* m1, const float* v1,
                        const float* g2, const float* b2, const float* m2, const float* v2,
                        float* a1, float* s1, float* a2, float* s2)
{
    int i = blockIdx.x * 256 + threadIdx.x;
    if (i < 512) { float a = g1[i] * rsqrtf(v1[i] + EPSF); a1[i] = a; s1[i] = b1[i] - m1[i] * a; }
    if (i < 256) { float a = g2[i] * rsqrtf(v2[i] + EPSF); a2[i] = a; s2[i] = b2[i] - m2[i] * a; }
}

extern "C" void kernel_launch(void* const* d_in, const int* in_sizes, int n_in,
                              void* d_out, int out_size, void* d_ws, size_t ws_size,
                              hipStream_t stream)
{
    const float* x      = (const float*)d_in[0];
    const float* qkv_w  = (const float*)d_in[1];
    const float* qkv_b  = (const float*)d_in[2];
    const float* fc_w   = (const float*)d_in[3];
    const float* fc_b   = (const float*)d_in[4];
    const float* ln_g   = (const float*)d_in[5];
    const float* ln_b   = (const float*)d_in[6];
    const float* proj_w = (const float*)d_in[7];
    const float* proj_b = (const float*)d_in[8];
    const float* w1     = (const float*)d_in[9];
    const float* bn1_g  = (const float*)d_in[10];
    const float* bn1_b  = (const float*)d_in[11];
    const float* bn1_m  = (const float*)d_in[12];
    const float* bn1_v  = (const float*)d_in[13];
    const float* w2     = (const float*)d_in[14];
    const float* bn2_g  = (const float*)d_in[15];
    const float* bn2_b  = (const float*)d_in[16];
    const float* bn2_m  = (const float*)d_in[17];
    const float* bn2_v  = (const float*)d_in[18];
    float* out = (float*)d_out;

    // workspace: region0 (157MB) hosts qkv -> normed -> h1 (lifetimes disjoint)
    //            region1 (52MB)  hosts stacked -> x1   (lifetimes disjoint)
    char* ws = (char*)d_ws;
    const size_t r0sz = (size_t)B_ * 768 * HW * 4;     // 157.3 MB
    const size_t r1sz = (size_t)B_ * 256 * HW * 4;     //  52.4 MB
    float* qkv_buf = (float*)(ws);                     // region0
    float* normed  = (float*)(ws);                     // region0 (after attn)
    float* h1      = (float*)(ws);                     // region0 (after proj)
    float* stacked = (float*)(ws + r0sz);              // region1
    float* x1      = (float*)(ws + r0sz);              // region1 (after fuse_ln)
    float* a1 = (float*)(ws + r0sz + r1sz);
    float* s1 = a1 + 512;
    float* a2 = s1 + 512;
    float* s2 = a2 + 256;

    bn_prep<<<2, 256, 0, stream>>>(bn1_g, bn1_b, bn1_m, bn1_v,
                                   bn2_g, bn2_b, bn2_m, bn2_v, a1, s1, a2, s2);

    // qkv = x @ qkv_w^T + qkv_b                      [B,768,HW]
    gemm_wx<<<dim3(50, 6, 8), 256, 0, stream>>>(x, qkv_w, nullptr, qkv_b,
                                                nullptr, qkv_buf, 256, 768, 0);
    // 4 dilated neighborhood-attention branches  ->  stacked [B,4,64,HW]
    attn_kernel<<<dim3(25, 4, 8), 256, 0, stream>>>(qkv_buf, stacked);
    // scale-mix + residual + LayerNorm(64)       ->  normed  [B,256,HW]
    fuse_ln_kernel<<<dim3(25, 1, 8), 256, 0, stream>>>(stacked, fc_w, fc_b,
                                                       ln_g, ln_b, normed);
    // x1 = x + normed @ proj_w^T + proj_b            [B,256,HW]
    gemm_wx<<<dim3(50, 2, 8), 256, 0, stream>>>(normed, proj_w, nullptr, proj_b,
                                                x, x1, 256, 256, 0);
    // h = silu(bn1(x1 @ w1^T))                       [B,512,HW]
    gemm_wx<<<dim3(50, 4, 8), 256, 0, stream>>>(x1, w1, a1, s1,
                                                nullptr, h1, 256, 512, 1);
    // out = x1 + bn2(h @ w2^T)                       [B,256,HW]
    gemm_wx<<<dim3(50, 2, 8), 256, 0, stream>>>(h1, w2, a2, s2,
                                                x1, out, 512, 256, 0);
}
